// CTRNN_57002805952616
// MI455X (gfx1250) — matrix operand order repaired and checked by reading, not compiled, not run on hardware
//
// CTRNN forward for MI455X (gfx1250).
//   out = [output (S,B,H) | hidden (B,H) | input_projection (S,B,H)] fp32
// Pipeline:
//   0) one-time fp32->bf16 conversion of W_in / W_hh into workspace
//   1) ip = x @ W_in^T + b_in : bf16 WMMA GEMM, ping-pong LDS tiles,
//      B staged via GLOBAL_LOAD_ASYNC_TO_LDS_B128 (ASYNCcnt)
//   2) persistent 512-step scan (16 blocks, atomic grid barrier). fp32 h state
//      lives in registers; only the bf16 shadow of h round-trips global memory
//      as the next step's A operand. Async-to-LDS staging, 1 barrier/K-iter.
#include <hip/hip_runtime.h>

#define S_LEN 512
#define BATCH 128
#define IN_DIM 512
#define HID 1024
#define ALPHA 0.2f
#define BH (BATCH * HID)

typedef __attribute__((ext_vector_type(16))) __bf16 v16bf;
typedef __attribute__((ext_vector_type(8)))  float  v8f;
typedef __attribute__((ext_vector_type(2)))  float  v2f;
typedef __attribute__((ext_vector_type(2)))  __bf16 v2bf;

union BF16x16 { uint4 q[2]; v16bf v; };

// two fp32 -> one dword of packed bf16 (lowers to v_cvt_pk_bf16_f32)
__device__ __forceinline__ unsigned packcvt(float a, float b) {
    v2f f = {a, b};
    union { v2bf h; unsigned u; } x;
    x.h = __builtin_convertvector(f, v2bf);
    return x.u;
}

// LDS byte offset of a __shared__ object (low 32 bits of generic address)
__device__ __forceinline__ unsigned ldsAddr(const void* p) {
    return (unsigned)(unsigned long long)p;
}

// gfx1250 async copy: 16 bytes global -> LDS per lane, tracked by ASYNCcnt
__device__ __forceinline__ void asyncLdsB128(unsigned lds, unsigned long long g) {
    asm volatile("global_load_async_to_lds_b128 %0, %1, off"
                 :: "v"(lds), "v"(g) : "memory");
}
__device__ __forceinline__ void waitAsync() {
    asm volatile("s_wait_asynccnt 0x0" ::: "memory");
}

// LDS row stride in dwords: 20 (=80B, 16B-aligned, conflict-free for 16 rows)
#define LDSTR 20
#define ABUF (128 * LDSTR)   // one A tile buffer, dwords
#define BBUF (64 * LDSTR)    // one B tile buffer, dwords

// ---------------------------------------------------------------------------
// One-time fp32 -> packed bf16 conversion (8 elements / thread)
// ---------------------------------------------------------------------------
__global__ __launch_bounds__(256) void cvt_f32_bf16(
    const float* __restrict__ s, unsigned* __restrict__ d, int n8)
{
    int i = blockIdx.x * 256 + threadIdx.x;
    if (i < n8) {
        const float4* sp = (const float4*)(s + (size_t)i * 8);
        float4 f0 = sp[0], f1 = sp[1];
        uint4 o;
        o.x = packcvt(f0.x, f0.y); o.y = packcvt(f0.z, f0.w);
        o.z = packcvt(f1.x, f1.y); o.w = packcvt(f1.z, f1.w);
        ((uint4*)d)[i] = o;
    }
}

// ---------------------------------------------------------------------------
// Kernel 1: ip[m][n] = sum_k x[m][k]*W_in[n][k] + b_in[n]
// Block tile 128(M) x 64(N), K-step 32, 8 waves x (16x64 of C each).
// ---------------------------------------------------------------------------
__global__ __launch_bounds__(256) void ctrnn_ip_gemm(
    const float* __restrict__ x,        // (65536, 512) fp32
    const __bf16* __restrict__ Wb,      // (1024, 512) bf16
    const float* __restrict__ b_in,     // (1024)
    float* __restrict__ ip)             // (65536, 1024)
{
    __shared__ unsigned As[2 * ABUF];
    __shared__ unsigned Bs[2 * BBUF];

    const int tid  = threadIdx.x;
    const int lane = tid & 31;
    const int wave = tid >> 5;
    const int n0   = blockIdx.x * 64;
    const int m0   = blockIdx.y * 128;

    const int arow  = tid >> 1;   // 0..127
    const int ahalf = tid & 1;
    const int brow  = tid >> 2;   // 0..63
    const int bseg  = tid & 3;

    const int am = wave * 16 + (lane & 15);
    const int bn = lane & 15;
    const int ha = (lane >> 4) * 4;  // A: dwords {ha..ha+3, 8+ha..8+ha+3}
    const int hb = (lane >> 4) * 4;  // B: dwords {hb..hb+3, 8+hb..8+hb+3}

    const unsigned long long wbG =
        (unsigned long long)Wb + ((size_t)(n0 + brow) * IN_DIM + bseg * 8) * 2;
    const unsigned bLdsStage = ldsAddr(Bs) + (brow * LDSTR + bseg * 4) * 4;
    const unsigned aStage = arow * LDSTR + ahalf * 8;   // dword index in buffer
    const float* aSrc = x + (size_t)(m0 + arow) * IN_DIM + ahalf * 16;

    const unsigned aRd = am * LDSTR + ha;               // compute-side indices
    const unsigned bRd = bn * LDSTR + hb;

    v8f acc[4] = {};

    // prologue: stage k0 = 0 into buffer 0
    asyncLdsB128(bLdsStage, wbG);
    {
        const float4* s = (const float4*)aSrc;
        float4 f0 = s[0], f1 = s[1], f2 = s[2], f3 = s[3];
        unsigned* d = As + aStage;
        d[0] = packcvt(f0.x, f0.y); d[1] = packcvt(f0.z, f0.w);
        d[2] = packcvt(f1.x, f1.y); d[3] = packcvt(f1.z, f1.w);
        d[4] = packcvt(f2.x, f2.y); d[5] = packcvt(f2.z, f2.w);
        d[6] = packcvt(f3.x, f3.y); d[7] = packcvt(f3.z, f3.w);
    }

    for (int k0 = 0; k0 < IN_DIM; k0 += 32) {
        const int cur = (k0 >> 5) & 1;
        const int nxt = cur ^ 1;
        waitAsync();
        __syncthreads();   // tile[cur] fully staged for all waves

        const bool more = (k0 + 32) < IN_DIM;
        float4 f0, f1, f2, f3;
        if (more) {
            // overlap: async B copy + global A loads for the next tile
            asyncLdsB128(bLdsStage + nxt * (BBUF * 4), wbG + (k0 + 32) * 2);
            const float4* s = (const float4*)(aSrc + k0 + 32);
            f0 = s[0]; f1 = s[1]; f2 = s[2]; f3 = s[3];
        }

        // load all operands up front so ds_loads pipeline with counted waits
        BF16x16 a, b[4];
        a.q[0] = *(const uint4*)&As[cur * ABUF + aRd];
        a.q[1] = *(const uint4*)&As[cur * ABUF + aRd + 8];
#pragma unroll
        for (int ns = 0; ns < 4; ++ns) {
            b[ns].q[0] = *(const uint4*)&Bs[cur * BBUF + bRd + ns * 16 * LDSTR];
            b[ns].q[1] = *(const uint4*)&Bs[cur * BBUF + bRd + ns * 16 * LDSTR + 8];
        }
#pragma unroll
        for (int ns = 0; ns < 4; ++ns) {
            acc[ns] = __builtin_amdgcn_wmma_f32_16x16x32_bf16(
                false, a.v, false, b[ns].v, (short)0, acc[ns], false, false);
        }

        if (more) {
            unsigned* d = As + nxt * ABUF + aStage;
            d[0] = packcvt(f0.x, f0.y); d[1] = packcvt(f0.z, f0.w);
            d[2] = packcvt(f1.x, f1.y); d[3] = packcvt(f1.z, f1.w);
            d[4] = packcvt(f2.x, f2.y); d[5] = packcvt(f2.z, f2.w);
            d[6] = packcvt(f3.x, f3.y); d[7] = packcvt(f3.z, f3.w);
        }
    }

    const int half = lane >> 4;
#pragma unroll
    for (int ns = 0; ns < 4; ++ns) {
        const int n = n0 + ns * 16 + (lane & 15);
        const float bias = b_in[n];
#pragma unroll
        for (int r = 0; r < 8; ++r) {
            const int m = m0 + wave * 16 + half * 8 + r;
            ip[(size_t)m * HID + n] = acc[ns][r] + bias;
        }
    }
}

// ---------------------------------------------------------------------------
// Kernel 2: persistent scan, 16 blocks (64-wide N slice each), 512 steps with
// an atomic grid barrier. fp32 h state lives in registers; bf16 shadow of h
// (global, ping-pong) is the A operand; B = bf16 W_hh.
// ---------------------------------------------------------------------------
__global__ __launch_bounds__(256) void ctrnn_scan(
    const __bf16* __restrict__ Whb,   // (1024, 1024) bf16
    const float* __restrict__ b_hh,   // (1024)
    const float* __restrict__ ip,     // (512, 128, 1024)
    float* __restrict__ out,          // (512, 128, 1024)
    __bf16* __restrict__ hb0,         // (128, 1024) bf16 ping
    __bf16* __restrict__ hb1,         // (128, 1024) bf16 pong
    unsigned* __restrict__ bar)       // monotonic arrival counter (init 0)
{
    __shared__ unsigned As[2 * ABUF];
    __shared__ unsigned Bs[2 * BBUF];

    const int tid  = threadIdx.x;
    const int lane = tid & 31;
    const int wave = tid >> 5;
    const int n0   = blockIdx.x * 64;
    const unsigned NB = gridDim.x;

    const int arow  = tid >> 1;
    const int ahalf = tid & 1;
    const int brow  = tid >> 2;
    const int bseg  = tid & 3;

    const int am = wave * 16 + (lane & 15);
    const int bn = lane & 15;
    const int ha = (lane >> 4) * 4;
    const int hb = (lane >> 4) * 4;
    const int half = lane >> 4;

    const unsigned long long whG =
        (unsigned long long)Whb + ((size_t)(n0 + brow) * HID + bseg * 8) * 2;
    const unsigned long long aOff = ((size_t)arow * HID + ahalf * 16) * 2;
    const unsigned aLdsStage = ldsAddr(As) + (arow * LDSTR + ahalf * 8) * 4;
    const unsigned bLdsStage = ldsAddr(Bs) + (brow * LDSTR + bseg * 4) * 4;

    const unsigned aRd = am * LDSTR + ha;
    const unsigned bRd = bn * LDSTR + hb;

    // persistent fp32 hidden state: this lane's 32 owned elements of h
    v8f hreg[4] = {};

    for (int t = 0; t < S_LEN; ++t) {
        const __bf16* __restrict__ hbin  = (t & 1) ? hb1 : hb0;
        __bf16*       __restrict__ hbout = (t & 1) ? hb0 : hb1;
        const float*  __restrict__ ipt   = ip  + (size_t)t * BH;
        float*        __restrict__ outt  = out + (size_t)t * BH;

        const unsigned long long haG = (unsigned long long)hbin + aOff;

        v8f acc[4] = {};

        // prologue: stage k0 = 0 into buffer 0
        asyncLdsB128(aLdsStage, haG);
        asyncLdsB128(aLdsStage + 16, haG + 16);
        asyncLdsB128(bLdsStage, whG);

        for (int k0 = 0; k0 < HID; k0 += 32) {
            const int cur = (k0 >> 5) & 1;
            const int nxt = cur ^ 1;
            waitAsync();
            __syncthreads();   // tile[cur] staged for all waves

            if (k0 + 32 < HID) {   // overlap next tile's copies with WMMA
                asyncLdsB128(aLdsStage + nxt * (ABUF * 4), haG + (k0 + 32) * 2);
                asyncLdsB128(aLdsStage + nxt * (ABUF * 4) + 16,
                             haG + (k0 + 32) * 2 + 16);
                asyncLdsB128(bLdsStage + nxt * (BBUF * 4), whG + (k0 + 32) * 2);
            }

            BF16x16 a, b[4];
            a.q[0] = *(const uint4*)&As[cur * ABUF + aRd];
            a.q[1] = *(const uint4*)&As[cur * ABUF + aRd + 8];
#pragma unroll
            for (int ns = 0; ns < 4; ++ns) {
                b[ns].q[0] =
                    *(const uint4*)&Bs[cur * BBUF + bRd + ns * 16 * LDSTR];
                b[ns].q[1] =
                    *(const uint4*)&Bs[cur * BBUF + bRd + ns * 16 * LDSTR + 8];
            }
#pragma unroll
            for (int ns = 0; ns < 4; ++ns) {
                acc[ns] = __builtin_amdgcn_wmma_f32_16x16x32_bf16(
                    false, a.v, false, b[ns].v, (short)0, acc[ns], false, false);
            }
        }

        // epilogue: h_new = relu(acc + ip + b); h = 0.8*h + 0.2*h_new
#pragma unroll
        for (int ns = 0; ns < 4; ++ns) {
            const int n = n0 + ns * 16 + (lane & 15);
            const float bias = b_hh[n];
#pragma unroll
            for (int r = 0; r < 8; ++r) {
                const int m = wave * 16 + half * 8 + r;
                const size_t idx = (size_t)m * HID + n;
                float pre = acc[ns][r] + ipt[idx] + bias;
                float hn  = fmaxf(pre, 0.0f);
                float ho  = (1.0f - ALPHA) * hreg[ns][r] + ALPHA * hn;
                hreg[ns][r] = ho;
                hbout[idx]  = (__bf16)ho;
                __builtin_nontemporal_store(ho, &outt[idx]);
            }
        }

        // ---- grid barrier (monotonic counter, device scope) ----
        __threadfence();
        __syncthreads();
        if (tid == 0) {
            __hip_atomic_fetch_add(bar, 1u, __ATOMIC_RELEASE,
                                   __HIP_MEMORY_SCOPE_AGENT);
            const unsigned target = NB * (unsigned)(t + 1);
            while (__hip_atomic_load(bar, __ATOMIC_ACQUIRE,
                                     __HIP_MEMORY_SCOPE_AGENT) < target) {
                __builtin_amdgcn_s_sleep(1);
            }
        }
        __syncthreads();
    }
}

// ---------------------------------------------------------------------------
extern "C" void kernel_launch(void* const* d_in, const int* in_sizes, int n_in,
                              void* d_out, int out_size, void* d_ws, size_t ws_size,
                              hipStream_t stream) {
    (void)in_sizes; (void)n_in; (void)out_size; (void)ws_size;

    const float* x    = (const float*)d_in[0];
    const float* W_in = (const float*)d_in[1];
    const float* b_in = (const float*)d_in[2];
    const float* W_hh = (const float*)d_in[3];
    const float* b_hh = (const float*)d_in[4];

    float* out        = (float*)d_out;
    float* out_output = out;                                  // S*B*H
    float* out_hidden = out + (size_t)S_LEN * BH;             // B*H
    float* out_ip     = out_hidden + (size_t)BH;              // S*B*H

    // workspace layout
    __bf16*   hb0   = (__bf16*)d_ws;                // BH bf16
    __bf16*   hb1   = hb0 + (size_t)BH;             // BH bf16
    __bf16*   Wi_bf = hb1 + (size_t)BH;             // 1024*512 bf16
    __bf16*   Wh_bf = Wi_bf + (size_t)HID * IN_DIM; // 1024*1024 bf16
    unsigned* bar   = (unsigned*)(Wh_bf + (size_t)HID * HID);

    // h(0) = 0 (bf16 ping buffer); barrier counter = 0
    hipMemsetAsync(hb0, 0, (size_t)BH * sizeof(__bf16), stream);
    hipMemsetAsync(bar, 0, sizeof(unsigned), stream);

    // 0) one-time weight conversion to bf16
    cvt_f32_bf16<<<dim3((HID * IN_DIM / 8) / 256), 256, 0, stream>>>(
        W_in, (unsigned*)Wi_bf, HID * IN_DIM / 8);
    cvt_f32_bf16<<<dim3((HID * HID / 8) / 256), 256, 0, stream>>>(
        W_hh, (unsigned*)Wh_bf, HID * HID / 8);

    // 1) input projection into the ip output slice (also feeds the scan)
    dim3 gIp(HID / 64, (S_LEN * BATCH) / 128);  // (16, 512)
    ctrnn_ip_gemm<<<gIp, 256, 0, stream>>>(x, Wi_bf, b_in, out_ip);

    // 2) persistent 512-step scan, 16 co-resident blocks
    ctrnn_scan<<<dim3(HID / 64), 256, 0, stream>>>(
        Wh_bf, b_hh, out_ip, out_output, hb0, hb1, bar);

    // 3) hidden = output[S-1] (h state never leaves registers in fp32)
    hipMemcpyAsync(out_hidden, out_output + (size_t)(S_LEN - 1) * BH,
                   (size_t)BH * sizeof(float), hipMemcpyDeviceToDevice, stream);
}
